// RNN_69337952026849
// MI455X (gfx1250) — matrix operand order repaired and checked by reading, not compiled, not run on hardware
//
#include <hip/hip_runtime.h>
#include <hip/hip_bf16.h>

#define T_STEPS 32768
#define IN_DIM  512
#define H_DIM   1024
#define NWG     16
#define ROWS    (H_DIM / NWG)   // 64 rows of W_h per workgroup

typedef __attribute__((ext_vector_type(16))) __bf16        v16bf;
typedef __attribute__((ext_vector_type(8)))  float         v8f;
typedef __attribute__((ext_vector_type(4)))  unsigned int  v4u;
typedef __attribute__((ext_vector_type(2)))  unsigned int  v2u;
typedef __attribute__((ext_vector_type(4)))  float         v4f;
typedef __attribute__((ext_vector_type(4)))  unsigned short v4s;

union Frag { v4u u4[2]; v16bf v; };

__device__ __forceinline__ unsigned short f2bf(float f) {
    unsigned int u = __float_as_uint(f);
    u += 0x7FFFu + ((u >> 16) & 1u);   // round-to-nearest-even
    return (unsigned short)(u >> 16);
}
__device__ __forceinline__ float bf2f(unsigned short s) {
    return __uint_as_float(((unsigned int)s) << 16);
}

// ---------------- fp32 -> bf16 conversion (vectorized) ----------------
__global__ __launch_bounds__(256) void cvt_bf16_kernel(const float* __restrict__ src,
                                                       unsigned short* __restrict__ dst,
                                                       int n4) {
    int i = blockIdx.x * 256 + threadIdx.x;
    if (i >= n4) return;
    v4f f = *(const v4f*)(src + (size_t)i * 4);
    v4s s;
    s.x = f2bf(f.x); s.y = f2bf(f.y); s.z = f2bf(f.z); s.w = f2bf(f.w);
    *(v4s*)(dst + (size_t)i * 4) = s;
}

// ---------------- init: zero h double-buffer + barrier counter ----------------
__global__ __launch_bounds__(256) void init_kernel(float* __restrict__ hbuf,
                                                   unsigned int* __restrict__ ctr) {
    for (int i = threadIdx.x; i < 2 * H_DIM; i += 256) hbuf[i] = 0.0f;
    if (threadIdx.x == 0) *ctr = 0u;
}

// ---------------- Phase 1: xin = x @ W_in.T + (b_in + b_h), bf16 WMMA ----------------
// A = x [T x IN] row-major bf16; B = W_in.T, i.e. W_in [H x IN] row-major bf16
// (both have K contiguous). One 16x16 output tile per wave, 8 waves per block.
__global__ __launch_bounds__(256) void gemm_xin_kernel(const unsigned short* __restrict__ xbf,
                                                       const unsigned short* __restrict__ winbf,
                                                       const float* __restrict__ b_in,
                                                       const float* __restrict__ b_h,
                                                       unsigned short* __restrict__ xinbf) {
    const int wave = threadIdx.x >> 5;
    const int lane = threadIdx.x & 31;
    const int gt   = blockIdx.x * 8 + wave;          // global tile id
    const int tn   = gt & 63;                        // 1024/16 = 64 N tiles
    const int tm   = gt >> 6;                        // 32768/16 = 2048 M tiles

    const int mloc  = lane & 15;                     // M index for A rows / N index for B cols
    const int half  = lane >> 4;                     // lane group 0/1
    const int arow  = tm * 16 + mloc;                // this lane's A row
    const int bcol  = tn * 16 + mloc;                // this lane's B column (W_in row)
    const int akoff = half * 16;                     // A: lanes16-31 hold K=16..31
    const int bkoff = half * 16;                     // B: lanes16-31 hold K=16..31

    v8f c = {0.f, 0.f, 0.f, 0.f, 0.f, 0.f, 0.f, 0.f};

    const unsigned short* aBase = xbf   + (size_t)arow * IN_DIM;
    const unsigned short* bBase = winbf + (size_t)bcol * IN_DIM;

    for (int kc = 0; kc < IN_DIM; kc += 32) {
        Frag a, b;
        // A fragment: this lane holds K[akoff..akoff+15] contiguous
        // (same k order as B)  -> two dwordx4 loads
        a.u4[0] = *(const v4u*)(aBase + kc + akoff);
        a.u4[1] = *(const v4u*)(aBase + kc + akoff + 8);
        // B fragment (16-bit B 32x16): this lane holds K[bkoff..bkoff+15] contiguous
        b.u4[0] = *(const v4u*)(bBase + kc + bkoff);
        b.u4[1] = *(const v4u*)(bBase + kc + bkoff + 8);

        c = __builtin_amdgcn_wmma_f32_16x16x32_bf16(
                /*neg_a=*/false, a.v, /*neg_b=*/false, b.v,
                /*c_mod=*/(short)0, c, /*reuse_a=*/false, /*reuse_b=*/false);
    }

    const float bias = b_in[bcol] + b_h[bcol];
    // D layout: VGPR v -> row M = v + half*8, col N = lane&15
    #pragma unroll
    for (int v = 0; v < 8; ++v) {
        int m = v + half * 8;
        xinbf[(size_t)(tm * 16 + m) * H_DIM + bcol] = f2bf(c[v] + bias);
    }
}

// ---------------- Phase 2: persistent 16-WG recurrence scan ----------------
// Each WG keeps a 64x1024 bf16 slice of W_h in LDS for all 32768 steps.
// h double-buffered in global (L2); one agent-scope atomic barrier per step.
__global__ __launch_bounds__(256) void rnn_scan_kernel(const unsigned short* __restrict__ xinbf,
                                                       const float* __restrict__ W_h,
                                                       float* __restrict__ hbuf,
                                                       unsigned int* __restrict__ ctr,
                                                       float* __restrict__ out) {
    extern __shared__ char smem_raw[];
    float*          hs  = (float*)smem_raw;                      // 1024 f32 = 4 KB
    unsigned short* wls = (unsigned short*)(smem_raw + H_DIM * 4); // 64*1024 bf16 = 128 KB

    const int tid  = threadIdx.x;
    const int wg   = blockIdx.x;          // 0..15
    const int row  = tid >> 2;            // 0..63 (local W_h row)
    const int part = tid & 3;             // K quarter
    const int kb   = part * (H_DIM / 4);  // 0,256,512,768
    const int grow = wg * ROWS + row;     // global hidden index this thread owns (part==0)

    // Load this WG's contiguous 64x1024 fp32 W_h slice, convert to bf16 in LDS.
    {
        const v4f* src = (const v4f*)(W_h + (size_t)wg * ROWS * H_DIM);
        for (int i = tid; i < ROWS * H_DIM / 4; i += 256) {
            v4f w4 = src[i];
            v4s s;
            s.x = f2bf(w4.x); s.y = f2bf(w4.y); s.z = f2bf(w4.z); s.w = f2bf(w4.w);
            *(v4s*)(wls + (size_t)i * 4) = s;
        }
    }
    __syncthreads();

    for (int t = 0; t < T_STEPS; ++t) {
        // Stage h_{t-1} (buffer t&1) from L2 into LDS.
        const float* hsrc = hbuf + (t & 1) * H_DIM;
        for (int i = tid; i < H_DIM; i += 256)
            hs[i] = __hip_atomic_load(hsrc + i, __ATOMIC_ACQUIRE, __HIP_MEMORY_SCOPE_AGENT);
        __syncthreads();

        // 1/4 of one dot product per thread, vectorized LDS reads.
        float sum = 0.f;
        const unsigned short* wr = wls + (size_t)row * H_DIM + kb;
        const float*          hp = hs + kb;
        #pragma unroll 8
        for (int k = 0; k < H_DIM / 4; k += 4) {
            v2u w2 = *(const v2u*)(wr + k);   // 4 bf16
            v4f h4 = *(const v4f*)(hp + k);
            sum += bf2f((unsigned short)(w2.x      )) * h4.x;
            sum += bf2f((unsigned short)(w2.x >> 16)) * h4.y;
            sum += bf2f((unsigned short)(w2.y      )) * h4.z;
            sum += bf2f((unsigned short)(w2.y >> 16)) * h4.w;
        }
        // reduce the 4 K-quarters (adjacent lanes in one wave32)
        sum += __shfl_xor(sum, 1, 4);
        sum += __shfl_xor(sum, 2, 4);

        if (part == 0) {
            float a  = bf2f(xinbf[(size_t)t * H_DIM + grow]) + sum;
            float hn = tanhf(a);
            __hip_atomic_store(hbuf + ((t + 1) & 1) * H_DIM + grow, hn,
                               __ATOMIC_RELEASE, __HIP_MEMORY_SCOPE_AGENT);
            if (t == T_STEPS - 1) out[grow] = hn;
        }
        __syncthreads();

        // One device-wide barrier per step (monotonic counter; reset by init_kernel).
        if (tid == 0) {
            __hip_atomic_fetch_add(ctr, 1u, __ATOMIC_ACQ_REL, __HIP_MEMORY_SCOPE_AGENT);
            const unsigned int target = (unsigned int)(t + 1) * NWG;
            while (__hip_atomic_load(ctr, __ATOMIC_ACQUIRE, __HIP_MEMORY_SCOPE_AGENT) < target)
                __builtin_amdgcn_s_sleep(1);
        }
        __syncthreads();
    }
}

extern "C" void kernel_launch(void* const* d_in, const int* in_sizes, int n_in,
                              void* d_out, int out_size, void* d_ws, size_t ws_size,
                              hipStream_t stream) {
    (void)in_sizes; (void)n_in; (void)out_size; (void)ws_size;
    const float* x    = (const float*)d_in[0];   // [T, IN]
    const float* W_in = (const float*)d_in[1];   // [H, IN]
    const float* b_in = (const float*)d_in[2];   // [H]
    const float* W_h  = (const float*)d_in[3];   // [H, H]
    const float* b_h  = (const float*)d_in[4];   // [H]
    float*       out  = (float*)d_out;           // [H, 1]

    // Workspace layout (~102 MB):
    unsigned short* xbf   = (unsigned short*)d_ws;                 // T*IN  bf16 (32 MB)
    unsigned short* winbf = xbf   + (size_t)T_STEPS * IN_DIM;      // H*IN  bf16 ( 1 MB)
    unsigned short* xinbf = winbf + (size_t)H_DIM  * IN_DIM;       // T*H   bf16 (64 MB)
    float*          hbuf  = (float*)(xinbf + (size_t)T_STEPS * H_DIM); // 2*H f32 (double buffer)
    unsigned int*   ctr   = (unsigned int*)(hbuf + 2 * H_DIM);     // barrier counter

    // 1) convert x and W_in to bf16
    {
        int n4x = T_STEPS * IN_DIM / 4;
        cvt_bf16_kernel<<<(n4x + 255) / 256, 256, 0, stream>>>(x, xbf, n4x);
        int n4w = H_DIM * IN_DIM / 4;
        cvt_bf16_kernel<<<(n4w + 255) / 256, 256, 0, stream>>>(W_in, winbf, n4w);
    }
    // 2) zero h0 and barrier counter (must happen every call: counter is monotonic in-kernel)
    init_kernel<<<1, 256, 0, stream>>>(hbuf, ctr);

    // 3) WMMA GEMM: xin = x @ W_in.T + (b_in + b_h), stored bf16
    {
        int tiles  = (T_STEPS / 16) * (H_DIM / 16);   // 131072
        int blocks = tiles / 8;                       // 8 waves per block
        gemm_xin_kernel<<<blocks, 256, 0, stream>>>(xbf, winbf, b_in, b_h, xinbf);
    }

    // 4) persistent recurrence: 16 WGs, W_h slice in 128 KB LDS each (+4 KB h stage)
    {
        size_t lds = (size_t)H_DIM * 4 + (size_t)ROWS * H_DIM * 2;  // 135168 B
        rnn_scan_kernel<<<NWG, 256, lds, stream>>>(xinbf, W_h, hbuf, ctr, out);
    }
}
